// RSAGEConv2d_6150393168696
// MI455X (gfx1250) — matrix operand order repaired and checked by reading, not compiled, not run on hardware
//
#include <hip/hip_runtime.h>

typedef __attribute__((ext_vector_type(16))) _Float16 v16h;
typedef __attribute__((ext_vector_type(8)))  _Float16 v8h;
typedef __attribute__((ext_vector_type(4)))  _Float16 v4h;
typedef __attribute__((ext_vector_type(8)))  float    v8f;

constexpr int Bn = 8;     // batches
constexpr int Nn = 4096;  // nodes
constexpr int CI = 64;    // in channels
constexpr int CO = 64;    // out channels
constexpr int Kn = 16;    // neighbors

union HU { v16h v; v8h h[2]; };

// A-fragment loader for v_wmma_f32_16x16x32_f16, row-major source [rows x row_stride] halves.
// Contiguous K layout (matches B): lanes 0-15 hold K {0..15}, lanes 16-31 hold K {16..31}.
static __device__ __forceinline__ v16h load_a_frag(const _Float16* __restrict__ base,
                                                   int row_stride, int m0, int kk, int lane) {
  const int row = m0 + (lane & 15);
  const int k1  = kk + ((lane >> 4) << 4);
  const _Float16* p = base + (size_t)row * row_stride + k1;
  HU u;
  u.h[0] = *(const v8h*)(p);
  u.h[1] = *(const v8h*)(p + 8);
  return u.v;
}

// ---------------- prep: x [B,C,N] f32 -> xt [B,N,C] f16 (tiled transpose) ----------------
__global__ __launch_bounds__(256) void prep_xt(const float* __restrict__ x,
                                               _Float16* __restrict__ xt) {
  __shared__ float tile[32][33];
  const int nt = blockIdx.x * 32, ct = blockIdx.y * 32, b = blockIdx.z;
  for (int j = threadIdx.y; j < 32; j += 8)
    tile[j][threadIdx.x] = x[((size_t)b * CI + ct + j) * Nn + nt + threadIdx.x];
  __syncthreads();
  for (int j = threadIdx.y; j < 32; j += 8)
    xt[((size_t)b * Nn + nt + j) * CI + ct + threadIdx.x] = (_Float16)tile[threadIdx.x][j];
}

// ---------------- prep: weights f32 -> f16 ----------------
__global__ __launch_bounds__(256) void prep_w(const float* __restrict__ wp,
                                              const float* __restrict__ wn,
                                              _Float16* __restrict__ wph,
                                              _Float16* __restrict__ wnh) {
  const int t = blockIdx.x * 256 + threadIdx.x;
  if (t < CO * CI)        wph[t] = (_Float16)wp[t];
  if (t < CO * (CI + CO)) wnh[t] = (_Float16)wn[t];
}

// -------- fused gather + GEMM1 + maxK (in-register) + concat + GEMM2 + bias + normalize ----
__global__ __launch_bounds__(256) void rsage_main(const _Float16* __restrict__ xt,   // [B,N,CI] f16
                                                  const _Float16* __restrict__ wpre, // [CO,CI] f16
                                                  const _Float16* __restrict__ wnn,  // [CO,CI+CO] f16
                                                  const float*    __restrict__ bias, // [CO]
                                                  const long long* __restrict__ eidx,// [2,B,N,K], plane 0
                                                  float* __restrict__ out) {         // [B,CO,N]
  constexpr int XG_STRIDE  = 72;   // 64 + 8 pad halves -> optimal 2-way b128 LDS access
  constexpr int CAT_STRIDE = 136;  // 128 + 8 pad halves
  constexpr int OUT_STRIDE = 68;   // 64 + 4 pad floats

  __shared__ __align__(16) _Float16 sXg[256 * XG_STRIDE];     // gathered [k][node][64 ch]
  __shared__ __align__(16) _Float16 sCat[16 * CAT_STRIDE];    // [node][x(64) | aggr(64)]
  __shared__ __align__(16) float    sPart[2 * 16 * OUT_STRIDE]; // per-k-half partial max [node][ch]
  __shared__ __align__(16) float    sOut[16 * OUT_STRIDE];    // [node][64 ch]
  __shared__ float sRn[16];

  const int t    = threadIdx.x;
  const int lane = t & 31;
  const int wave = t >> 5;
  const int n0   = blockIdx.x * 16;
  const int b    = blockIdx.y;

  // ---- Phase A: async-gather one 128B neighbor column per thread straight into LDS ----
  {
    const int nl = t >> 4;   // node in tile
    const int k  = t & 15;   // neighbor
    const long long idx = eidx[((size_t)b * Nn + (n0 + nl)) * Kn + k];
    const _Float16* src = xt + ((size_t)b * Nn + (int)idx) * CI;
    _Float16* dst = &sXg[(k * 16 + nl) * XG_STRIDE];   // [k][node] layout
    const unsigned           ldsoff = (unsigned)(uintptr_t)dst;
    const unsigned long long gaddr  = (unsigned long long)(uintptr_t)src;
    // CDNA5 async DMA: global -> LDS, no VGPR round-trip, tracked by ASYNCcnt.
    asm volatile("global_load_async_to_lds_b128 %0, %1, off offset:0"  :: "v"(ldsoff), "v"(gaddr) : "memory");
    asm volatile("global_load_async_to_lds_b128 %0, %1, off offset:16" :: "v"(ldsoff), "v"(gaddr) : "memory");
    asm volatile("global_load_async_to_lds_b128 %0, %1, off offset:32" :: "v"(ldsoff), "v"(gaddr) : "memory");
    asm volatile("global_load_async_to_lds_b128 %0, %1, off offset:48" :: "v"(ldsoff), "v"(gaddr) : "memory");
    asm volatile("global_load_async_to_lds_b128 %0, %1, off offset:64" :: "v"(ldsoff), "v"(gaddr) : "memory");
    asm volatile("global_load_async_to_lds_b128 %0, %1, off offset:80" :: "v"(ldsoff), "v"(gaddr) : "memory");
    asm volatile("global_load_async_to_lds_b128 %0, %1, off offset:96" :: "v"(ldsoff), "v"(gaddr) : "memory");
    asm volatile("global_load_async_to_lds_b128 %0, %1, off offset:112" :: "v"(ldsoff), "v"(gaddr) : "memory");

    // stage x tile (f16) into cat buffer: 8 bytes per thread
    const _Float16* xsrc = xt + ((size_t)b * Nn + (n0 + nl)) * CI + k * 4;
    *(v4h*)&sCat[nl * CAT_STRIDE + k * 4] = *(const v4h*)xsrc;
  }
  asm volatile("s_wait_asynccnt 0" ::: "memory");
  __syncthreads();

  // ---- GEMM1: acc_k = wpre @ x_jk per neighbor; running element-wise max (relu via init 0) ----
  {
    const int rt    = wave & 3;   // channel row-tile
    const int khalf = wave >> 2;  // neighbors [khalf*8, khalf*8+8)

    v16h a[2];
#pragma unroll
    for (int ks = 0; ks < 2; ++ks)
      a[ks] = load_a_frag(wpre, CI, rt * 16, ks * 32, lane);

    float mx[8] = {0.f, 0.f, 0.f, 0.f, 0.f, 0.f, 0.f, 0.f};  // = relu identity
#pragma unroll
    for (int k = 0; k < 8; ++k) {
      const int kk = khalf * 8 + k;
      v8f acc = {};
#pragma unroll
      for (int ks = 0; ks < 2; ++ks) {
        // B 32x16 fragment: N = node (lane%16), lane half selects contiguous 16 K values
        const _Float16* bp = &sXg[(kk * 16 + (lane & 15)) * XG_STRIDE
                                  + ks * 32 + ((lane >> 4) << 4)];
        HU u; u.h[0] = *(const v8h*)bp; u.h[1] = *(const v8h*)(bp + 8);
        acc = __builtin_amdgcn_wmma_f32_16x16x32_f16(false, a[ks], false, u.v,
                                                     (short)0, acc, false, false);
      }
#pragma unroll
      for (int r = 0; r < 8; ++r) mx[r] = fmaxf(mx[r], acc[r]);
    }
    // D layout: lanes 0-15 -> node=lane, ch=rt*16+r ; lanes 16-31 -> node=lane-16, ch=+8
    float* pp = &sPart[(khalf * 16 + (lane & 15)) * OUT_STRIDE + rt * 16 + ((lane >> 4) << 3)];
#pragma unroll
    for (int r = 0; r < 8; ++r) pp[r] = mx[r];   // contiguous -> ds_store_b128 x2
  }
  __syncthreads();

  // ---- combine the two neighbor-half maxima, write aggr half into cat ----
#pragma unroll
  for (int i = 0; i < 4; ++i) {
    const int idx  = t + i * 256;      // 1024 = 16 nodes x 64 ch
    const int node = idx >> 6;
    const int c    = idx & 63;
    const float v = fmaxf(sPart[node * OUT_STRIDE + c],
                          sPart[(16 + node) * OUT_STRIDE + c]);
    sCat[node * CAT_STRIDE + CI + c] = (_Float16)v;
  }
  __syncthreads();

  // ---- GEMM2: relu(wnn @ cat) + bias; waves 0..3 each own one 16-channel row tile ----
  if (wave < 4) {
    const int rt = wave;
    v8f acc = {};
#pragma unroll
    for (int ks = 0; ks < 4; ++ks) {
      const v16h afr = load_a_frag(wnn, CI + CO, rt * 16, ks * 32, lane);
      const _Float16* bp = &sCat[(lane & 15) * CAT_STRIDE + ks * 32 + ((lane >> 4) << 4)];
      HU u; u.h[0] = *(const v8h*)bp; u.h[1] = *(const v8h*)(bp + 8);
      acc = __builtin_amdgcn_wmma_f32_16x16x32_f16(false, afr, false, u.v,
                                                   (short)0, acc, false, false);
    }
    const int node = lane & 15;
    const int chb  = rt * 16 + ((lane >> 4) << 3);
#pragma unroll
    for (int r = 0; r < 8; ++r)
      sOut[node * OUT_STRIDE + chb + r] = fmaxf(acc[r], 0.0f) + bias[chb + r];
  }
  __syncthreads();

  // ---- per-node L2 norm over channels ----
  if (t < 16) {
    float s = 0.0f;
#pragma unroll
    for (int c = 0; c < CO; ++c) { const float v = sOut[t * OUT_STRIDE + c]; s += v * v; }
    sRn[t] = 1.0f / fmaxf(sqrtf(s), 1e-12f);
  }
  __syncthreads();

  // ---- coalesced store: out[b, c, n0+node] ----
#pragma unroll
  for (int i = 0; i < 4; ++i) {
    const int idx  = t + i * 256;
    const int c    = idx >> 4;
    const int node = idx & 15;
    out[((size_t)b * CO + c) * Nn + n0 + node] = sOut[node * OUT_STRIDE + c] * sRn[node];
  }
}

extern "C" void kernel_launch(void* const* d_in, const int* in_sizes, int n_in,
                              void* d_out, int out_size, void* d_ws, size_t ws_size,
                              hipStream_t stream) {
  const float*     x     = (const float*)d_in[0];
  // d_in[1] = x_0 : unused by the reference computation
  const float*     w_pre = (const float*)d_in[2];
  const float*     w_nn  = (const float*)d_in[3];
  const float*     bias  = (const float*)d_in[4];
  const long long* eidx  = (const long long*)d_in[5];
  float* out = (float*)d_out;

  _Float16* xt   = (_Float16*)d_ws;                  // B*N*CI halves = 4 MB
  _Float16* wpre = xt + (size_t)Bn * Nn * CI;
  _Float16* wnn  = wpre + CO * CI;

  prep_xt<<<dim3(Nn / 32, CI / 32, Bn), dim3(32, 8), 0, stream>>>(x, xt);
  prep_w<<<32, 256, 0, stream>>>(w_pre, w_nn, wpre, wnn);
  rsage_main<<<dim3(Nn / 16, Bn), 256, 0, stream>>>(xt, wpre, wnn, bias, eidx, out);
}
